// SpatiallyModulatedAttention_66151086293334
// MI455X (gfx1250) — matrix operand order repaired and checked by reading, not compiled, not run on hardware
//
#include <hip/hip_runtime.h>
#include <cstdint>
#include <cstddef>

typedef _Float16 h16 __attribute__((ext_vector_type(16)));
typedef _Float16 h8  __attribute__((ext_vector_type(8)));
typedef float    f8  __attribute__((ext_vector_type(8)));

#define S_LEN  1024
#define BATCH  4
#define DMODEL 1024
#define NHEAD  16
#define DK     64
#define MROWS  (BATCH * S_LEN)   // 4096

// ---- WMMA fragment helpers (A/B k order contiguous, this file's choice; D per ISA) -------------------
// A (16x32 f16): lane<16 -> row=lane,    K=k0..k0+15 contiguous
//                lane>=16 -> row=lane-16, K=k0+16..k0+31 contiguous
// B (32x16 f16): lane<16 -> col=lane,    K=k0..k0+15 contiguous; lane>=16 -> +16
// D (16x16 f32): col = lane&15, rows = (lane>>4)*8 + j

__device__ __forceinline__ h16 ld_frag(const _Float16* p0, const _Float16* p1) {
    h8 lo = *(const h8*)p0;
    h8 hi = *(const h8*)p1;
    return __builtin_shufflevector(lo, hi, 0,1,2,3,4,5,6,7,8,9,10,11,12,13,14,15);
}

__device__ __forceinline__ f8 wmma_f16(h16 a, h16 b, f8 c) {
    return __builtin_amdgcn_wmma_f32_16x16x32_f16(false, a, false, b, (short)0, c,
                                                  false, false);
}

__device__ __forceinline__ f8 f8_zero() {
    f8 z = {0.f,0.f,0.f,0.f,0.f,0.f,0.f,0.f};
    return z;
}

// ---- CDNA5 async global->LDS copy (ASYNCcnt-tracked, no VGPR data path) -----
__device__ __forceinline__ void async_ld_b128(const void* gsrc, void* ldst) {
    unsigned d = (unsigned)(size_t)ldst;             // LDS byte offset
    asm volatile("global_load_async_to_lds_b128 %0, %1, off"
                 :: "v"(d), "v"((unsigned long long)(size_t)gsrc)
                 : "memory");
}
__device__ __forceinline__ void async_wait0() {
    asm volatile("s_wait_asynccnt 0x0" ::: "memory");
}

// ---- Kernel 1: input projections  Y = X @ W^T -------------------------------
// X: [S,B,D] f32, W: [Dout,Din] f32.  mode 0 -> [B,H,S,DK] f16 (Q scaled, K)
// mode 2 -> transposed [B,H,DK,S] f16 (V).
__global__ __launch_bounds__(128) void k_proj(
    const float* __restrict__ X, const float* __restrict__ W,
    _Float16* __restrict__ out, int mode, float scale)
{
    const int tid = threadIdx.x, wave = tid >> 5, lane = tid & 31;
    const int r = lane & 15, hs = lane >> 4;
    const int m0 = blockIdx.x * 64;
    const int n0 = blockIdx.y * 64;
    const int b  = m0 >> 10;
    const int s0 = m0 & 1023;
    __shared__ __align__(16) _Float16 As[64][40];
    __shared__ __align__(16) _Float16 Bs[64][40];
    f8 acc[4] = {f8_zero(), f8_zero(), f8_zero(), f8_zero()};

    for (int k0 = 0; k0 < DMODEL; k0 += 32) {
        __syncthreads();
        for (int i = tid; i < 512; i += 128) {
            int row = i >> 3, c4 = (i & 7) << 2;
            float4 f = *(const float4*)&X[((size_t)(s0 + row) * BATCH + b) * DMODEL + k0 + c4];
            As[row][c4+0] = (_Float16)f.x; As[row][c4+1] = (_Float16)f.y;
            As[row][c4+2] = (_Float16)f.z; As[row][c4+3] = (_Float16)f.w;
        }
        for (int i = tid; i < 512; i += 128) {
            int row = i >> 3, c4 = (i & 7) << 2;
            float4 f = *(const float4*)&W[(size_t)(n0 + row) * DMODEL + k0 + c4];
            Bs[row][c4+0] = (_Float16)f.x; Bs[row][c4+1] = (_Float16)f.y;
            Bs[row][c4+2] = (_Float16)f.z; Bs[row][c4+3] = (_Float16)f.w;
        }
        __syncthreads();
        h16 a = ld_frag(&As[wave*16 + r][hs*16], &As[wave*16 + r][hs*16 + 8]);
        #pragma unroll
        for (int jt = 0; jt < 4; ++jt) {
            h16 bf = ld_frag(&Bs[jt*16 + r][hs*16], &Bs[jt*16 + r][hs*16 + 8]);
            acc[jt] = wmma_f16(a, bf, acc[jt]);
        }
    }
    const int h = n0 >> 6;
    #pragma unroll
    for (int jt = 0; jt < 4; ++jt) {
        int dk = jt*16 + r;
        #pragma unroll
        for (int j = 0; j < 8; ++j) {
            int s = s0 + wave*16 + hs*8 + j;
            float v = acc[jt][j] * scale;
            if (mode == 2)
                out[(((size_t)b*NHEAD + h)*DK + dk)*S_LEN + s] = (_Float16)v;
            else
                out[(((size_t)b*NHEAD + h)*S_LEN + s)*DK + dk] = (_Float16)v;
        }
    }
}

// ---- Kernel 2: fused logits*sph -> softmax -> p_attn out -> P@V^T -----------
// One block per (q-block of 64, bh). 256 threads = 8 waves.
// Full 64x1024 f32 logits stripe lives in LDS; K/V tiles are streamed with
// double-buffered async global->LDS copies overlapped with WMMA.
__global__ __launch_bounds__(256) void k_attn(
    const _Float16* __restrict__ qh,   // [B*H, S, DK]
    const _Float16* __restrict__ kh,   // [B*H, S, DK]
    const _Float16* __restrict__ vt,   // [B*H, DK, S]
    const float* __restrict__ sph,     // [B*H, S, S]
    float* __restrict__ P,             // [B*H, S, S]  (p_attn output)
    _Float16* __restrict__ ctx)        // [B*S, D]
{
    const int tid  = threadIdx.x;
    const int wave = tid >> 5, lane = tid & 31;
    const int r = lane & 15, hs = lane >> 4;
    const int mw = wave & 3, nw = wave >> 2;   // 4 m-strips x 2 n-halves
    const int q0 = blockIdx.x * 64;
    const int bh = blockIdx.y;
    const int b = bh >> 4, h = bh & 15;

    __shared__ __align__(16) float    Ls[64][1044];    // 267,264 B logits stripe
    __shared__ __align__(16) _Float16 Qs[64][72];      //   9,216 B
    __shared__ __align__(16) _Float16 Ks[2][128][72];  //  36,864 B (K, then V)

    // Q tile (async) + first K chunk (async, buffer 0)
    for (int i = tid; i < 512; i += 256) {
        int row = i >> 3, c8 = (i & 7) << 3;
        async_ld_b128(&qh[((size_t)bh*S_LEN + q0 + row)*DK + c8], &Qs[row][c8]);
    }
    for (int i = tid; i < 1024; i += 256) {
        int row = i >> 3, c8 = (i & 7) << 3;
        async_ld_b128(&kh[((size_t)bh*S_LEN + row)*DK + c8], &Ks[0][row][c8]);
    }
    async_wait0();
    __syncthreads();

    // ---- phase 1: logits = (Q Kt) * sph  -> Ls ------------------------------
    for (int ic = 0; ic < 8; ++ic) {
        const int nc = ic * 128, bb = ic & 1;
        if (ic < 7) {                                  // prefetch next K chunk
            for (int i = tid; i < 1024; i += 256) {
                int row = i >> 3, c8 = (i & 7) << 3;
                async_ld_b128(&kh[((size_t)bh*S_LEN + nc + 128 + row)*DK + c8],
                              &Ks[bb ^ 1][row][c8]);
            }
        }
        f8 acc[4] = {f8_zero(), f8_zero(), f8_zero(), f8_zero()};
        #pragma unroll
        for (int k0 = 0; k0 < DK; k0 += 32) {
            h16 a = ld_frag(&Qs[mw*16 + r][k0 + hs*16], &Qs[mw*16 + r][k0 + hs*16 + 8]);
            #pragma unroll
            for (int jt = 0; jt < 4; ++jt) {
                h16 bf = ld_frag(&Ks[bb][nw*64 + jt*16 + r][k0 + hs*16],
                                 &Ks[bb][nw*64 + jt*16 + r][k0 + hs*16 + 8]);
                acc[jt] = wmma_f16(a, bf, acc[jt]);
            }
        }
        #pragma unroll
        for (int jt = 0; jt < 4; ++jt) {
            int kc = nc + nw*64 + jt*16 + r;
            #pragma unroll
            for (int j = 0; j < 8; ++j) {
                int mrow = mw*16 + hs*8 + j;
                Ls[mrow][kc] = acc[jt][j] *
                               sph[((size_t)bh*S_LEN + q0 + mrow)*S_LEN + kc];
            }
        }
        if (ic < 7) async_wait0();
        __syncthreads();
    }

    // Kick off V chunk 0 (buffer 0) so its latency hides under the softmax.
    {
        int row = tid >> 2, c8 = (tid & 3) << 3;
        async_ld_b128(&vt[((size_t)bh*DK + row)*S_LEN + c8], &Ks[0][row][c8]);
    }

    // ---- phase 2: row softmax; write p_attn; pack f16 P into Ls in place ----
    _Float16* Lh = (_Float16*)&Ls[0][0];
    const int LHS = 1044 * 2;                  // halves per row
    for (int rr = 0; rr < 8; ++rr) {
        int row = wave*8 + rr;
        float mx = -3.0e38f;
        #pragma unroll
        for (int i = 0; i < 32; ++i) mx = fmaxf(mx, Ls[row][lane + 32*i]);
        #pragma unroll
        for (int off = 16; off > 0; off >>= 1) mx = fmaxf(mx, __shfl_xor(mx, off));
        float pv[32];
        float sm = 0.f;
        #pragma unroll
        for (int i = 0; i < 32; ++i) {
            pv[i] = __expf(Ls[row][lane + 32*i] - mx);
            sm += pv[i];
        }
        #pragma unroll
        for (int off = 16; off > 0; off >>= 1) sm += __shfl_xor(sm, off);
        float inv = 1.0f / sm;
        #pragma unroll
        for (int i = 0; i < 32; ++i) {
            int c = lane + 32*i;
            float p = pv[i] * inv;
            P[((size_t)bh*S_LEN + q0 + row)*S_LEN + c] = p;
            Lh[row*LHS + c] = (_Float16)p;     // writes trail reads (deps via pv)
        }
    }
    async_wait0();
    __syncthreads();

    // ---- phase 3: ctx = P @ V^T  (A-fragments straight from LDS-resident P) -
    f8 acc2[2] = {f8_zero(), f8_zero()};
    for (int ik = 0; ik < 32; ++ik) {
        const int k0 = ik * 32, bb = ik & 1;
        if (ik < 31) {                                 // prefetch next V chunk
            int row = tid >> 2, c8 = (tid & 3) << 3;
            async_ld_b128(&vt[((size_t)bh*DK + row)*S_LEN + k0 + 32 + c8],
                          &Ks[bb ^ 1][row][c8]);
        }
        h16 a = ld_frag(&Lh[(mw*16 + r)*LHS + k0 + hs*16],
                        &Lh[(mw*16 + r)*LHS + k0 + hs*16 + 8]);
        #pragma unroll
        for (int jt = 0; jt < 2; ++jt) {
            h16 bf = ld_frag(&Ks[bb][nw*32 + jt*16 + r][hs*16],
                             &Ks[bb][nw*32 + jt*16 + r][hs*16 + 8]);
            acc2[jt] = wmma_f16(a, bf, acc2[jt]);
        }
        if (ik < 31) { async_wait0(); __syncthreads(); }
    }
    #pragma unroll
    for (int jt = 0; jt < 2; ++jt) {
        int dk = nw*32 + jt*16 + r;
        #pragma unroll
        for (int j = 0; j < 8; ++j) {
            int qi = q0 + mw*16 + hs*8 + j;
            ctx[((size_t)b*S_LEN + qi)*DMODEL + h*DK + dk] = (_Float16)acc2[jt][j];
        }
    }
}

// ---- Kernel 3: output projection + residual ---------------------------------
__global__ __launch_bounds__(128) void k_oproj(
    const _Float16* __restrict__ ctx,  // [B*S, D]
    const float* __restrict__ Wo,      // [D, D]
    const float* __restrict__ qin,     // [S, B, D] residual source
    float* __restrict__ preln)         // [B*S, D]
{
    const int tid = threadIdx.x, wave = tid >> 5, lane = tid & 31;
    const int r = lane & 15, hs = lane >> 4;
    const int m0 = blockIdx.x * 64;
    const int n0 = blockIdx.y * 64;
    const int b = m0 >> 10, s0 = m0 & 1023;
    __shared__ __align__(16) _Float16 As[64][40];
    __shared__ __align__(16) _Float16 Bs[64][40];
    f8 acc[4] = {f8_zero(), f8_zero(), f8_zero(), f8_zero()};

    for (int k0 = 0; k0 < DMODEL; k0 += 32) {
        __syncthreads();
        for (int i = tid; i < 256; i += 128) {
            int row = i >> 2, c8 = (i & 3) << 3;
            *(h8*)&As[row][c8] = *(const h8*)&ctx[(size_t)(m0 + row)*DMODEL + k0 + c8];
        }
        for (int i = tid; i < 512; i += 128) {
            int row = i >> 3, c4 = (i & 7) << 2;
            float4 f = *(const float4*)&Wo[(size_t)(n0 + row)*DMODEL + k0 + c4];
            Bs[row][c4+0] = (_Float16)f.x; Bs[row][c4+1] = (_Float16)f.y;
            Bs[row][c4+2] = (_Float16)f.z; Bs[row][c4+3] = (_Float16)f.w;
        }
        __syncthreads();
        h16 a = ld_frag(&As[wave*16 + r][hs*16], &As[wave*16 + r][hs*16 + 8]);
        #pragma unroll
        for (int jt = 0; jt < 4; ++jt) {
            h16 bf = ld_frag(&Bs[jt*16 + r][hs*16], &Bs[jt*16 + r][hs*16 + 8]);
            acc[jt] = wmma_f16(a, bf, acc[jt]);
        }
    }
    #pragma unroll
    for (int jt = 0; jt < 4; ++jt) {
        int o = n0 + jt*16 + r;
        #pragma unroll
        for (int j = 0; j < 8; ++j) {
            int s = s0 + wave*16 + hs*8 + j;
            int m = m0 + wave*16 + hs*8 + j;
            preln[(size_t)m*DMODEL + o] = acc[jt][j] + qin[((size_t)s*BATCH + b)*DMODEL + o];
        }
    }
}

// ---- Kernel 4: LayerNorm over last dim, write [S,B,D] -----------------------
__global__ __launch_bounds__(256) void k_ln(
    const float* __restrict__ preln,   // [B*S, D]
    const float* __restrict__ gamma,
    const float* __restrict__ beta,
    float* __restrict__ out)           // [S, B, D]
{
    const int tid = threadIdx.x;
    const int m = blockIdx.x;
    const int b = m >> 10, s = m & 1023;
    __shared__ float r1[256], r2[256];
    float4 x = *(const float4*)&preln[(size_t)m*DMODEL + tid*4];
    r1[tid] = x.x + x.y + x.z + x.w;
    r2[tid] = x.x*x.x + x.y*x.y + x.z*x.z + x.w*x.w;
    __syncthreads();
    for (int off = 128; off > 0; off >>= 1) {
        if (tid < off) { r1[tid] += r1[tid+off]; r2[tid] += r2[tid+off]; }
        __syncthreads();
    }
    float mu  = r1[0] * (1.0f/DMODEL);
    float var = r2[0] * (1.0f/DMODEL) - mu*mu;
    float inv = rsqrtf(var + 1e-6f);
    float4 g  = *(const float4*)&gamma[tid*4];
    float4 be = *(const float4*)&beta[tid*4];
    float4 y;
    y.x = (x.x - mu)*inv*g.x + be.x;
    y.y = (x.y - mu)*inv*g.y + be.y;
    y.z = (x.z - mu)*inv*g.z + be.z;
    y.w = (x.w - mu)*inv*g.w + be.w;
    *(float4*)&out[((size_t)s*BATCH + b)*DMODEL + tid*4] = y;
}

// ---- Host launcher ----------------------------------------------------------
extern "C" void kernel_launch(void* const* d_in, const int* in_sizes, int n_in,
                              void* d_out, int out_size, void* d_ws, size_t ws_size,
                              hipStream_t stream) {
    (void)in_sizes; (void)n_in; (void)out_size; (void)ws_size;
    const float* q     = (const float*)d_in[0];
    const float* k     = (const float*)d_in[1];
    const float* v     = (const float*)d_in[2];
    const float* sph   = (const float*)d_in[3];
    const float* Wq    = (const float*)d_in[4];
    const float* Wk    = (const float*)d_in[5];
    const float* Wv    = (const float*)d_in[6];
    const float* Wo    = (const float*)d_in[7];
    const float* gamma = (const float*)d_in[8];
    const float* beta  = (const float*)d_in[9];

    float* out = (float*)d_out;                          // [S,B,D]   (output 0)
    float* P   = out + (size_t)S_LEN*BATCH*DMODEL;       // [B,H,S,S] (output 1)

    const size_t HE = (size_t)BATCH*NHEAD*S_LEN*DK;      // 4,194,304 halves
    _Float16* qh  = (_Float16*)d_ws;                     // [B,H,S,DK]
    _Float16* kh  = qh + HE;                             // [B,H,S,DK]
    _Float16* vt  = kh + HE;                             // [B,H,DK,S]
    _Float16* ctx = vt + HE;                             // [B*S, D]
    float*  preln = (float*)(ctx + (size_t)MROWS*DMODEL);// [B*S, D]

    dim3 bp(128);
    k_proj <<<dim3(64,16,1), bp, 0, stream>>>(q, Wq, qh, 0, 0.125f);  // 1/sqrt(64)
    k_proj <<<dim3(64,16,1), bp, 0, stream>>>(k, Wk, kh, 0, 1.0f);
    k_proj <<<dim3(64,16,1), bp, 0, stream>>>(v, Wv, vt, 2, 1.0f);
    k_attn <<<dim3(16,64,1), dim3(256), 0, stream>>>(qh, kh, vt, sph, P, ctx);
    k_oproj<<<dim3(64,16,1), bp, 0, stream>>>(ctx, Wo, q, preln);
    k_ln   <<<dim3(MROWS),   dim3(256), 0, stream>>>(preln, gamma, beta, out);
}
